// CantorTransformerBlock_40381282517088
// MI455X (gfx1250) — matrix operand order repaired and checked by reading, not compiled, not run on hardware
//
#include <hip/hip_runtime.h>
#include <hip/hip_bf16.h>

// ---------------------------------------------------------------------------
// Types for CDNA5 WMMA (wave32): v16bf A/B operands, v8f f32 accumulator.
// ---------------------------------------------------------------------------
typedef __bf16 bf16x16 __attribute__((ext_vector_type(16)));
typedef float  f32x8   __attribute__((ext_vector_type(8)));

union WFrag {
    bf16x16      v;
    unsigned int u[8];
    uint4        q[2];
};

__device__ __forceinline__ unsigned short f2bf_u(float f) {
    unsigned int u = __float_as_uint(f);
    u += 0x7fffu + ((u >> 16) & 1u);     // round-to-nearest-even
    return (unsigned short)(u >> 16);
}

__device__ __forceinline__ float gelu_exact(float x) {
    return 0.5f * x * (1.0f + erff(x * 0.70710678118654752f));
}

// ---------------------------------------------------------------------------
// CDNA5 async global->LDS staging (ASYNCcnt) with portable fallback.
// Builtin expects (v4i addrspace(1)*, v4i addrspace(3)*, imm offset, imm cpol).
// ---------------------------------------------------------------------------
#if defined(__has_builtin)
#if __has_builtin(__builtin_amdgcn_global_load_async_to_lds_b128)
#define HAVE_ASYNC_COPY 1
#endif
#endif
#ifndef HAVE_ASYNC_COPY
#define HAVE_ASYNC_COPY 0
#endif

typedef int v4i_vec __attribute__((__vector_size__(4 * sizeof(int))));
typedef __attribute__((address_space(1))) v4i_vec* gv4i_p;
typedef __attribute__((address_space(3))) v4i_vec* lv4i_p;

__device__ __forceinline__ void async_copy_b128(const unsigned short* gsrc,
                                                unsigned short* ldst) {
#if HAVE_ASYNC_COPY
    __builtin_amdgcn_global_load_async_to_lds_b128(
        (gv4i_p)gsrc, (lv4i_p)ldst, 0, 0);
#else
    *(uint4*)ldst = *(const uint4*)gsrc;
#endif
}

__device__ __forceinline__ void wait_async_le3() {
#if HAVE_ASYNC_COPY
#if __has_builtin(__builtin_amdgcn_s_wait_asynccnt)
    __builtin_amdgcn_s_wait_asynccnt(3);
#else
    asm volatile("s_wait_asynccnt 3" ::: "memory");
#endif
#endif
}

__device__ __forceinline__ void wait_async_le0() {
#if HAVE_ASYNC_COPY
#if __has_builtin(__builtin_amdgcn_s_wait_asynccnt)
    __builtin_amdgcn_s_wait_asynccnt(0);
#else
    asm volatile("s_wait_asynccnt 0" ::: "memory");
#endif
#endif
}

// ---------------------------------------------------------------------------
// fp32 [K][N]  ->  bf16 [N][K]  (weights, transposed so GEMM B-fragments are
// contiguous b128 LDS reads). 32x32 tiles via LDS, block = 256 threads.
// ---------------------------------------------------------------------------
__global__ __launch_bounds__(256)
void cvt_transpose_bf16(const float* __restrict__ in,      // K x N
                        unsigned short* __restrict__ out,  // N x K
                        int K, int N) {
    __shared__ float t[32][33];
    const int bk = blockIdx.y * 32, bn = blockIdx.x * 32;
    const int tx = threadIdx.x & 31, ty = threadIdx.x >> 5;  // 32 x 8
#pragma unroll
    for (int i = 0; i < 32; i += 8)
        t[ty + i][tx] = in[(size_t)(bk + ty + i) * N + bn + tx];
    __syncthreads();
#pragma unroll
    for (int i = 0; i < 32; i += 8)
        out[(size_t)(bn + ty + i) * K + bk + tx] = f2bf_u(t[tx][ty + i]);
}

// ---------------------------------------------------------------------------
// LayerNorm over D=512, writes bf16 activations for the WMMA GEMMs.
// One block (256 threads) per row; 2 elements/thread.
// ---------------------------------------------------------------------------
__global__ __launch_bounds__(256)
void layernorm512_bf16(const float* __restrict__ x,
                       const float* __restrict__ g,
                       const float* __restrict__ b,
                       unsigned short* __restrict__ out) {
    const int row = blockIdx.x;
    const int tid = threadIdx.x;
    const float* xr = x + (size_t)row * 512;
    float v0 = xr[tid], v1 = xr[tid + 256];

    __shared__ float red[256];
    red[tid] = v0 + v1;
    __syncthreads();
    for (int s = 128; s > 0; s >>= 1) {
        if (tid < s) red[tid] += red[tid + s];
        __syncthreads();
    }
    float mu = red[0] * (1.0f / 512.0f);
    __syncthreads();
    float d0 = v0 - mu, d1 = v1 - mu;
    red[tid] = d0 * d0 + d1 * d1;
    __syncthreads();
    for (int s = 128; s > 0; s >>= 1) {
        if (tid < s) red[tid] += red[tid + s];
        __syncthreads();
    }
    float rstd = rsqrtf(red[0] * (1.0f / 512.0f) + 1e-5f);
    size_t base = (size_t)row * 512;
    out[base + tid]       = f2bf_u(d0 * rstd * g[tid] + b[tid]);
    out[base + tid + 256] = f2bf_u(d1 * rstd * g[tid + 256] + b[tid + 256]);
}

// ---------------------------------------------------------------------------
// bf16 WMMA GEMM: C(MxN) = act(A(MxK) @ B(KxN) + bias) [+ resid]
//   B supplied TRANSPOSED (BT = N x K) so B-fragments are contiguous b128.
//   block = 256 threads = 8 waves; tile 128x64; each wave a 32x32 quadrant
//   as 2x2 v_wmma_f32_16x16x32_bf16 tiles; BK=32.
//   Double-buffered LDS, async global->LDS staging (3 async instr / wave /
//   tile -> s_wait_asynccnt 3 leaves only the prefetch outstanding).
// Fragment layouts per CDNA5 ISA 7.12.2 (wave32):
//   A: lane L holds M=L%16, K=(v>>2)*16+(L>>4)*8+(v&3)*2 (+1)  -> 2 x b128
//   B: lane L holds N=L%16, K=(v>>2)*16+(L>>4)*8+(v&3)*2 (+1)  -> 2 x b128
//   D: M=v+8*(L>>4), N=L%16
// ---------------------------------------------------------------------------
#define GEMM_BM 128
#define GEMM_BN 64
#define GEMM_BK 32
#define SA_STR  40   // ushorts: 80B rows keep all b128 accesses 16B-aligned
#define SBT_STR 40

__global__ __launch_bounds__(256)
void gemm_bf16_wmma(const unsigned short* __restrict__ A,   // M x K
                    const unsigned short* __restrict__ BT,  // N x K
                    const float* __restrict__ bias,
                    const float* __restrict__ resid,
                    float* __restrict__ Cf,
                    unsigned short* __restrict__ Cbf,
                    int M, int N, int K, int act) {
    __shared__ unsigned short sA[2][GEMM_BM * SA_STR];    // 2 x 10240 B
    __shared__ unsigned short sBT[2][GEMM_BN * SBT_STR];  // 2 x  5120 B

    const int tid  = threadIdx.x;
    const int lane = tid & 31;
    const int wave = tid >> 5;
    const int wm   = wave & 3;    // wave row (4)
    const int wn   = wave >> 2;   // wave col (2)
    const int bm0  = blockIdx.y * GEMM_BM;
    const int bn0  = blockIdx.x * GEMM_BN;

    // staging coordinates: A = 512 b128 chunks (2/thread), BT = 256 (1/thread)
    const int arow = tid >> 2;          // 0..63  (and +64)
    const int ac8  = (tid & 3) * 8;
    const int bcol = tid >> 2;          // 0..63
    const int bc8  = (tid & 3) * 8;

    auto stage = [&](int kt, int buf) {
        const int k0 = kt * GEMM_BK;
        async_copy_b128(A + (size_t)(bm0 + arow) * K + k0 + ac8,
                        &sA[buf][arow * SA_STR + ac8]);
        async_copy_b128(A + (size_t)(bm0 + 64 + arow) * K + k0 + ac8,
                        &sA[buf][(64 + arow) * SA_STR + ac8]);
        async_copy_b128(BT + (size_t)(bn0 + bcol) * K + k0 + bc8,
                        &sBT[buf][bcol * SBT_STR + bc8]);
    };

    f32x8 acc[2][2];
#pragma unroll
    for (int i = 0; i < 2; ++i)
#pragma unroll
        for (int j = 0; j < 2; ++j)
#pragma unroll
            for (int e = 0; e < 8; ++e) acc[i][j][e] = 0.0f;

    const int am = lane & 15;
    const int ag = lane >> 4;
    const int bn = lane & 15;
    const int bg = lane >> 4;

    const int NT = K / GEMM_BK;
    stage(0, 0);

    for (int kt = 0; kt < NT; ++kt) {
        const int buf = kt & 1;
        if (kt + 1 < NT) {
            stage(kt + 1, buf ^ 1);   // prev-iter end barrier made buf^1 free
            wait_async_le3();         // current tile's 3 async ops complete
        } else {
            wait_async_le0();
        }
        __syncthreads();              // all waves' current tile visible

        const unsigned short* pa = sA[buf];
        const unsigned short* pb = sBT[buf];

        WFrag fa[2], fb[2];
#pragma unroll
        for (int mt = 0; mt < 2; ++mt) {
            const unsigned short* base = pa + (wm * 32 + mt * 16 + am) * SA_STR + ag * 8;
            fa[mt].q[0] = *(const uint4*)(base);
            fa[mt].q[1] = *(const uint4*)(base + 16);
        }
#pragma unroll
        for (int nt = 0; nt < 2; ++nt) {
            const unsigned short* base = pb + (wn * 32 + nt * 16 + bn) * SBT_STR + bg * 8;
            fb[nt].q[0] = *(const uint4*)(base);
            fb[nt].q[1] = *(const uint4*)(base + 16);
        }

#pragma unroll
        for (int mt = 0; mt < 2; ++mt)
#pragma unroll
            for (int nt = 0; nt < 2; ++nt)
                acc[mt][nt] = __builtin_amdgcn_wmma_f32_16x16x32_bf16(
                    false, fa[mt].v, false, fb[nt].v,
                    (short)0, acc[mt][nt], false, false);
        __syncthreads();              // done reading buf before it is reused
    }

    // ---- epilogue ----
#pragma unroll
    for (int mt = 0; mt < 2; ++mt) {
#pragma unroll
        for (int nt = 0; nt < 2; ++nt) {
            int col   = bn0 + wn * 32 + nt * 16 + (lane & 15);
            int rbase = bm0 + wm * 32 + mt * 16 + (lane >> 4) * 8;
            float bcolv = bias[col];
#pragma unroll
            for (int v = 0; v < 8; ++v) {
                int row = rbase + v;
                float val = acc[mt][nt][v] + bcolv;
                if (act) val = gelu_exact(val);
                if (resid) val += resid[(size_t)row * N + col];
                if (Cf)  Cf[(size_t)row * N + col] = val;
                if (Cbf) Cbf[(size_t)row * N + col] = f2bf_u(val);
            }
        }
    }
}

// ---------------------------------------------------------------------------
// Cantor sparse attention: one wave per (query row, head); lane owns 4 of the
// 128 gathered keys. fp32 VALU dot products / wave softmax / shuffle PV.
// qkv layout: [2048][1536], q|k|v at col offsets 0|512|1024, head h at +64h.
// ---------------------------------------------------------------------------
__device__ __forceinline__ int key_index(int s, int j) {
    int idx;
    if (j < 64)       idx = s + j - 32;                // local window
    else if (j < 102) idx = s + (j - 83) * 64;         // medium stride
    else              return ((j - 102) * 2047) / 25;  // global linspace
    idx = idx < 0 ? 0 : idx;
    return idx > 2047 ? 2047 : idx;
}

__global__ __launch_bounds__(256)
void cantor_attention(const float* __restrict__ qkv,
                      const unsigned char* __restrict__ mask,
                      unsigned short* __restrict__ outbf) {
    const int lane = threadIdx.x & 31;
    const int wave = threadIdx.x >> 5;
    const int s = blockIdx.x * 8 + wave;
    const int h = blockIdx.y;

    const float4* q4 = (const float4*)(qkv + (size_t)s * 1536 + h * 64);

    float sc[4];
    int   kidx[4];
#pragma unroll
    for (int t = 0; t < 4; ++t) {
        int j  = lane + t * 32;
        int ki = key_index(s, j);
        kidx[t] = ki;
        const float4* k4 = (const float4*)(qkv + (size_t)ki * 1536 + 512 + h * 64);
        float acc = 0.0f;
#pragma unroll
        for (int d = 0; d < 16; ++d) {
            float4 qv = q4[d], kv = k4[d];
            acc += qv.x * kv.x + qv.y * kv.y + qv.z * kv.z + qv.w * kv.w;
        }
        sc[t] = mask[ki] ? acc * 0.125f : -3.4028235e38f;
    }

    float mx = fmaxf(fmaxf(sc[0], sc[1]), fmaxf(sc[2], sc[3]));
#pragma unroll
    for (int off = 16; off > 0; off >>= 1) mx = fmaxf(mx, __shfl_xor(mx, off));
    float aw[4], sum = 0.0f;
#pragma unroll
    for (int t = 0; t < 4; ++t) { aw[t] = __expf(sc[t] - mx); sum += aw[t]; }
#pragma unroll
    for (int off = 16; off > 0; off >>= 1) sum += __shfl_xor(sum, off);
    float inv = 1.0f / sum;
#pragma unroll
    for (int t = 0; t < 4; ++t) aw[t] *= inv;

    const int d0 = lane * 2;
    float o0 = 0.0f, o1 = 0.0f;
    for (int j = 0; j < 128; ++j) {
        float aj = __shfl(aw[j >> 5], j & 31);
        int   ki = __shfl(kidx[j >> 5], j & 31);
        const float2 vv = *(const float2*)(qkv + (size_t)ki * 1536 + 1024 + h * 64 + d0);
        o0 += aj * vv.x;
        o1 += aj * vv.y;
    }
    size_t ob = (size_t)s * 512 + h * 64 + d0;
    outbf[ob]     = f2bf_u(o0);
    outbf[ob + 1] = f2bf_u(o1);
}

// ---------------------------------------------------------------------------
// Host orchestration
// ---------------------------------------------------------------------------
extern "C" void kernel_launch(void* const* d_in, const int* in_sizes, int n_in,
                              void* d_out, int out_size, void* d_ws, size_t ws_size,
                              hipStream_t stream) {
    const float*         x     = (const float*)d_in[0];
    const unsigned char* mask  = (const unsigned char*)d_in[1];
    const float*         ln1_g = (const float*)d_in[2];
    const float*         ln1_b = (const float*)d_in[3];
    const float*         ln2_g = (const float*)d_in[4];
    const float*         ln2_b = (const float*)d_in[5];
    const float*         wqkv  = (const float*)d_in[6];
    const float*         bqkv  = (const float*)d_in[7];
    const float*         wo    = (const float*)d_in[8];
    const float*         bo    = (const float*)d_in[9];
    const float*         w1    = (const float*)d_in[10];
    const float*         b1    = (const float*)d_in[11];
    const float*         w2    = (const float*)d_in[12];
    const float*         b2    = (const float*)d_in[13];
    float* out = (float*)d_out;

    const int S = 2048, D = 512, F = 2048, NQKV = 1536;

    char*  ws  = (char*)d_ws;
    size_t off = 0;
    auto alloc = [&](size_t bytes) -> void* {
        off = (off + 255) & ~(size_t)255;
        void* p = ws + off;
        off += bytes;
        return p;
    };
    unsigned short* wqkv_t = (unsigned short*)alloc((size_t)D * NQKV * 2); // [NQKV][D]
    unsigned short* wo_t   = (unsigned short*)alloc((size_t)D * D * 2);    // [D][D]
    unsigned short* w1_t   = (unsigned short*)alloc((size_t)D * F * 2);    // [F][D]
    unsigned short* w2_t   = (unsigned short*)alloc((size_t)F * D * 2);    // [D][F]
    unsigned short* hln_bf = (unsigned short*)alloc((size_t)S * D * 2);
    float*          qkv_f  = (float*)alloc((size_t)S * NQKV * 4);
    unsigned short* attn_bf= (unsigned short*)alloc((size_t)S * D * 2);
    float*          xres   = (float*)alloc((size_t)S * D * 4);
    unsigned short* ln2_bf = (unsigned short*)alloc((size_t)S * D * 2);
    unsigned short* h1_bf  = (unsigned short*)alloc((size_t)S * F * 2);

    // 1) weights -> bf16, transposed to N-major for WMMA B fragments
    cvt_transpose_bf16<<<dim3(NQKV / 32, D / 32), 256, 0, stream>>>(wqkv, wqkv_t, D, NQKV);
    cvt_transpose_bf16<<<dim3(D / 32,    D / 32), 256, 0, stream>>>(wo,   wo_t,   D, D);
    cvt_transpose_bf16<<<dim3(F / 32,    D / 32), 256, 0, stream>>>(w1,   w1_t,   D, F);
    cvt_transpose_bf16<<<dim3(D / 32,    F / 32), 256, 0, stream>>>(w2,   w2_t,   F, D);

    // 2) LN1
    layernorm512_bf16<<<S, 256, 0, stream>>>(x, ln1_g, ln1_b, hln_bf);

    // 3) qkv = LN1(x) @ wqkv + bqkv
    gemm_bf16_wmma<<<dim3(NQKV / GEMM_BN, S / GEMM_BM), 256, 0, stream>>>(
        hln_bf, wqkv_t, bqkv, nullptr, qkv_f, nullptr, S, NQKV, D, 0);

    // 4) sparse attention -> bf16
    cantor_attention<<<dim3(S / 8, 8), 256, 0, stream>>>(qkv_f, mask, attn_bf);

    // 5) xres = x + attn @ wo + bo
    gemm_bf16_wmma<<<dim3(D / GEMM_BN, S / GEMM_BM), 256, 0, stream>>>(
        attn_bf, wo_t, bo, x, xres, nullptr, S, D, D, 0);

    // 6) LN2
    layernorm512_bf16<<<S, 256, 0, stream>>>(xres, ln2_g, ln2_b, ln2_bf);

    // 7) h1 = gelu(LN2 @ w1 + b1) -> bf16
    gemm_bf16_wmma<<<dim3(F / GEMM_BN, S / GEMM_BM), 256, 0, stream>>>(
        ln2_bf, w1_t, b1, nullptr, nullptr, h1_bf, S, F, D, 1);

    // 8) out = xres + h1 @ w2 + b2
    gemm_bf16_wmma<<<dim3(D / GEMM_BN, S / GEMM_BM), 256, 0, stream>>>(
        h1_bf, w2_t, b2, xres, out, nullptr, S, D, F, 0);
}
